// DNM_Conv_fold_69312182223549
// MI455X (gfx1250) — matrix operand order repaired and checked by reading, not compiled, not run on hardware
//
#include <hip/hip_runtime.h>
#include <hip/hip_bf16.h>

typedef __attribute__((ext_vector_type(16))) _Float16 v16h;
typedef __attribute__((ext_vector_type(8)))  _Float16 v8h;
typedef __attribute__((ext_vector_type(8)))  float    v8f;

// Problem constants (from reference setup_inputs)
#define BATCH 4
#define CH    64      // C (layernorm dim, GEMM K)
#define OCH   64      // O
#define NM    4       // M parallel convs
#define HH    384
#define WW    384
#define PIX_TILE 128  // pixels per workgroup
#define NROW  (NM*OCH)   // 256 stacked weight rows
#define HW    (HH*WW)

__global__ __launch_bounds__(128)
void dnm_conv_fold_kernel(const float* __restrict__ x,
                          const float* __restrict__ Wt,     // [M,O,C] row-major
                          const float* __restrict__ qp,
                          const float* __restrict__ gamma,
                          const float* __restrict__ beta,
                          float* __restrict__ out) {
    __shared__ __align__(32) _Float16 Wh[NROW * CH];       // [row][c]  32 KB
    __shared__ __align__(32) _Float16 Xh[PIX_TILE * CH];   // [p][c]    16 KB
    __shared__ float sG[CH];
    __shared__ float sB[CH];

    const int tid = threadIdx.x;

    // --- block -> (b, h, w0) ---
    const int bid   = blockIdx.x;
    const int tiles = WW / PIX_TILE;            // 3
    const int tw    = bid % tiles;
    const int h     = (bid / tiles) % HH;
    const int b     = bid / (tiles * HH);
    const int w0    = tw * PIX_TILE;

    // --- stage weights (f32 -> f16) into LDS, flat copy: Wh[row*64+c] ---
    for (int i = tid; i < NROW * CH; i += 128)
        Wh[i] = (_Float16)Wt[i];
    if (tid < CH) { sG[tid] = gamma[tid]; sB[tid] = beta[tid]; }
    __syncthreads();

    const float qv = qp[0];

    // --- LayerNorm: one pixel per thread, channels held in registers ---
    {
        const int p    = tid;                    // 0..127
        const int base = ((b * CH) * HH + h) * WW + w0 + p;
        float v[CH];
        float s = 0.f;
        #pragma unroll
        for (int c = 0; c < CH; ++c) {
            v[c] = x[base + c * HW];             // coalesced across threads
            s += v[c];
        }
        const float mu = s * (1.0f / CH);
        float s2 = 0.f;
        #pragma unroll
        for (int c = 0; c < CH; ++c) {
            const float d = v[c] - mu;
            s2 += d * d;
        }
        const float rs = rsqrtf(s2 * (1.0f / CH) + 1e-5f);
        #pragma unroll
        for (int c = 0; c < CH; ++c)
            Xh[p * CH + c] = (_Float16)((v[c] - mu) * rs * sG[c] + sB[c]);
    }
    __syncthreads();

    // --- WMMA phase: stacked [256x64] x [64x128] GEMM, fused relu/shift/M-sum ---
    const int wave = tid >> 5;                  // 0..3 -> owns oTile = wave
    const int lane = tid & 31;
    const int n16  = lane & 15;                 // column / row-in-tile selector
    const int g    = lane >> 4;                 // lane half

    // Preload ALL A fragments for this wave's oTile: 4 m x 2 k-steps (64 VGPRs),
    // reused across all 8 pixel tiles.
    // A layout: lane half g holds K = g*16..g*16+15 (per 32-k slab)
    const int oT = wave;
    v16h A[NM][2];
    #pragma unroll
    for (int m = 0; m < NM; ++m) {
        const int row = m * OCH + oT * 16 + n16;
        #pragma unroll
        for (int k = 0; k < 2; ++k) {
            A[m][k] = *(const v16h*)&Wh[row * CH + k * 32 + g * 16];
        }
    }

    // 8 pixel tiles per wave; steady-state loop: 4 ds B-loads -> 8 WMMAs -> epilogue
    for (int pT = 0; pT < 8; ++pT) {
        // B fragments (lane n=n16 holds K = g*16..g*16+15 of each 32-k slab)
        const int p = pT * 16 + n16;
        const v16h b0 = *(const v16h*)&Xh[p * CH +  0 + g * 16];
        const v16h b1 = *(const v16h*)&Xh[p * CH + 32 + g * 16];

        v8f acc = {};
        #pragma unroll
        for (int m = 0; m < NM; ++m) {
            v8f y = {};
            y = __builtin_amdgcn_wmma_f32_16x16x32_f16(
                    false, A[m][0], false, b0, (short)0, y, false, false);
            y = __builtin_amdgcn_wmma_f32_16x16x32_f16(
                    false, A[m][1], false, b1, (short)0, y, false, false);

            #pragma unroll
            for (int j = 0; j < 8; ++j)
                acc[j] += fmaxf(y[j] - qv, 0.0f);
        }

        // D layout: VGPR v -> row (v + 8*g), lane n16 -> column
        #pragma unroll
        for (int v = 0; v < 8; ++v) {
            const int o = oT * 16 + v + 8 * g;
            out[((b * OCH + o) * HH + h) * WW + w0 + pT * 16 + n16] = acc[v];
        }
    }
}

extern "C" void kernel_launch(void* const* d_in, const int* in_sizes, int n_in,
                              void* d_out, int out_size, void* d_ws, size_t ws_size,
                              hipStream_t stream) {
    const float* x     = (const float*)d_in[0];
    const float* Wt    = (const float*)d_in[1];
    const float* q     = (const float*)d_in[2];
    const float* gamma = (const float*)d_in[3];
    const float* beta  = (const float*)d_in[4];
    float* out = (float*)d_out;

    const int nBlocks = BATCH * HH * (WW / PIX_TILE);   // 4*384*3 = 4608
    dnm_conv_fold_kernel<<<nBlocks, 128, 0, stream>>>(x, Wt, q, gamma, beta, out);
}
